// Variational_DYConv2d_72318659330763
// MI455X (gfx1250) — matrix operand order repaired and checked by reading, not compiled, not run on hardware
//
#include <hip/hip_runtime.h>
#include <math.h>

// ---------------------------------------------------------------------------
// Variational dynamic conv2d for MI455X (gfx1250).
//   attn_inp -> folded into x during f32->bf16 conversion (scale_x_kernel)
//   attn_k   -> folded into the im2col B gather (scalar per 64-wide K-stage)
//   attn_oup -> f32 epilogue scale
// Main conv = per-batch implicit GEMM [COUT x (9*CIN)] x [(9*CIN) x HW],
// v_wmma_f32_16x16x32_bf16, double-buffered LDS pipeline with A tiles staged
// via global_load_async_to_lds_b128 (ASYNCcnt) overlapped with WMMA compute.
// ---------------------------------------------------------------------------

typedef __bf16 bf16;
typedef __attribute__((ext_vector_type(16))) __bf16 v16bf;
typedef __attribute__((ext_vector_type(8)))  float  v8f;

#define B_    32
#define CIN_  256
#define H_    56
#define W_    56
#define HW_   3136
#define COUT_ 256
#define HID_  16
#define KK_   9
#define KDIM_ 2304          // 9 taps * 256 channels, tap-major
#define KSTEP 64            // K per pipeline stage (2 WMMA K-steps)
#define LDS_STRIDE 72       // bf16 elems/row; 144B rows, 16B aligned, 36-bank skew
#define M_TILE 128
#define N_TILE 64

// ---------------------------- pooling --------------------------------------
__global__ void pool_kernel(const float* __restrict__ x, float* __restrict__ pooled) {
  __shared__ float red[256];
  const int bc = blockIdx.x;                       // b*CIN + c
  const float* p = x + (size_t)bc * HW_;
  float s = 0.f;
  for (int i = threadIdx.x; i < HW_; i += 256) s += p[i];
  red[threadIdx.x] = s;
  __syncthreads();
  for (int off = 128; off > 0; off >>= 1) {
    if (threadIdx.x < off) red[threadIdx.x] += red[threadIdx.x + off];
    __syncthreads();
  }
  if (threadIdx.x == 0) pooled[bc] = red[0] * (1.0f / HW_);
}

// ------------------------- tiny attention MLPs ------------------------------
__global__ void attn_kernel(const float* __restrict__ pooled,
                            const float* __restrict__ w_share, const float* __restrict__ b_share,
                            const float* __restrict__ w_inp,   const float* __restrict__ b_inp,
                            const float* __restrict__ w_oup,   const float* __restrict__ b_oup,
                            const float* __restrict__ w_k,     const float* __restrict__ b_k,
                            float* __restrict__ attn_inp, float* __restrict__ attn_oup,
                            float* __restrict__ attn_k) {
  __shared__ float hid[B_ * HID_];                 // 512 floats
  const int t = threadIdx.x;
  for (int idx = t; idx < B_ * HID_; idx += 256) {
    const int b = idx / HID_, h = idx % HID_;
    float s = b_share[h];
    for (int c = 0; c < CIN_; ++c) s += pooled[b * CIN_ + c] * w_share[c * HID_ + h];
    hid[idx] = fmaxf(s, 0.f);
  }
  __syncthreads();
  for (int idx = t; idx < B_ * CIN_; idx += 256) {
    const int b = idx / CIN_, c = idx % CIN_;
    float si = b_inp[c], so = b_oup[c];
    for (int h = 0; h < HID_; ++h) {
      const float hv = hid[b * HID_ + h];
      si += hv * w_inp[h * CIN_ + c];
      so += hv * w_oup[h * CIN_ + c];
    }
    attn_inp[idx] = 1.f / (1.f + __expf(-si));
    attn_oup[idx] = 1.f / (1.f + __expf(-so));
  }
  for (int idx = t; idx < B_ * KK_; idx += 256) {
    const int b = idx / KK_, k = idx % KK_;
    float s = b_k[k];
    for (int h = 0; h < HID_; ++h) s += hid[b * HID_ + h] * w_k[h * KK_ + k];
    attn_k[idx] = 1.f / (1.f + __expf(-s));
  }
}

// ------------- x * attn_inp -> bf16 (streaming, float4 loads) ---------------
__global__ void scale_x_kernel(const float* __restrict__ x,
                               const float* __restrict__ attn_inp,
                               bf16* __restrict__ xs) {
  const long long i4 = (long long)blockIdx.x * blockDim.x + threadIdx.x;
  const long long N4 = (long long)B_ * CIN_ * HW_ / 4;
  if (i4 >= N4) return;
  const int bc = (int)((i4 * 4) / HW_);            // HW_ % 4 == 0 -> one channel per float4
  const float sc = attn_inp[bc];
  const float4 v = ((const float4*)x)[i4];
  union { bf16 e[4]; unsigned long long u; } o;
  o.e[0] = (bf16)(v.x * sc);
  o.e[1] = (bf16)(v.y * sc);
  o.e[2] = (bf16)(v.z * sc);
  o.e[3] = (bf16)(v.w * sc);
  ((unsigned long long*)xs)[i4] = o.u;
}

// ------------- weight reorder: [o][i][tap] -> bf16 [o][tap*256+i] -----------
__global__ void wT_kernel(const float* __restrict__ weight, bf16* __restrict__ wT) {
  const int idx = blockIdx.x * blockDim.x + threadIdx.x;
  if (idx >= COUT_ * KDIM_) return;
  const int o = idx / KDIM_, r = idx % KDIM_;
  const int tap = r / CIN_, i = r % CIN_;
  wT[idx] = (bf16)weight[(o * CIN_ + i) * KK_ + tap];
}

// ----------------- async global -> LDS copy helpers (gfx1250) ---------------
__device__ __forceinline__ void async_copy_b128(unsigned lds_addr, const bf16* gptr) {
  // GLOBAL_LOAD_ASYNC_TO_LDS_B128, GV addressing: vdst = LDS byte address,
  // v[addr] = 64-bit global address.  Tracked by ASYNCcnt.
  asm volatile("global_load_async_to_lds_b128 %0, %1, off"
               :: "v"(lds_addr), "v"(gptr) : "memory");
}
__device__ __forceinline__ void wait_asynccnt0() {
#if __has_builtin(__builtin_amdgcn_s_wait_asynccnt)
  __builtin_amdgcn_s_wait_asynccnt(0);
#else
  asm volatile("s_wait_asynccnt 0x0" ::: "memory");
#endif
}

// --------------------------- main WMMA conv ---------------------------------
// grid = (49 spatial 8x8 patches, COUT/128, B), block = 256 (8 waves)
__global__ void __launch_bounds__(256)
dyconv_wmma_kernel(const bf16* __restrict__ xs, const bf16* __restrict__ wT,
                   const float* __restrict__ attn_k, const float* __restrict__ attn_oup,
                   float* __restrict__ out) {
  __shared__ bf16 As[2][M_TILE * LDS_STRIDE];      // 2 x 18 KB
  __shared__ bf16 Bs[2][N_TILE * LDS_STRIDE];      // 2 x  9 KB

  const int s     = blockIdx.x;                    // 0..48
  const int oBase = blockIdx.y * M_TILE;
  const int b     = blockIdx.z;
  const int hBase = (s / 7) * 8;
  const int wBase = (s % 7) * 8;

  const int t    = threadIdx.x;
  const int lane = t & 31;
  const int wave = t >> 5;                         // 0..7
  const int mWave = (wave >> 1) * 32;              // 0,32,64,96
  const int nWave = (wave &  1) * 32;              // 0,32

  // attn_k row for this batch: 9 uniform scalars, load once
  float akv[KK_];
#pragma unroll
  for (int k = 0; k < KK_; ++k) akv[k] = attn_k[b * KK_ + k];

  v8f acc[2][2];
#pragma unroll
  for (int mi = 0; mi < 2; ++mi)
#pragma unroll
    for (int ni = 0; ni < 2; ++ni)
#pragma unroll
      for (int e = 0; e < 8; ++e) acc[mi][ni][e] = 0.f;

  // A loader: thread copies 64B of one weight row (pure copy, async b128 x4)
  const int arow = t >> 1;                         // 0..127
  const int aseg = (t & 1) * 32;                   // element offset in 64-wide k
  const bf16* aSrcBase = wT + (size_t)(oBase + arow) * KDIM_ + aseg;
  unsigned aLds[2];
  aLds[0] = (unsigned)(unsigned long long)(const void*)&As[0][arow * LDS_STRIDE + aseg];
  aLds[1] = (unsigned)(unsigned long long)(const void*)&As[1][arow * LDS_STRIDE + aseg];

  // B loader: thread gathers 2 k-rows x 8 consecutive w positions
  const int bkk0 = (t >> 3) * 2;                   // 0,2,..,62
  const int bdh  = t & 7;                          // fixed image row within patch
  const int hRow = hBase + bdh;

  auto loadA = [&](int kb, int pp) {
    const bf16* src = aSrcBase + kb;
    const unsigned l = aLds[pp];
    async_copy_b128(l,      src);
    async_copy_b128(l + 16, src + 8);
    async_copy_b128(l + 32, src + 16);
    async_copy_b128(l + 48, src + 24);
  };

  auto loadB = [&](int kb, int pp) {
    const int tap = kb / CIN_;                     // stage never crosses a tap
    const int i0  = kb % CIN_;
    const int kh  = tap / 3 - 1;
    const int kw  = tap % 3 - 1;
    const float ak = akv[tap];
    const int hh  = hRow + kh;
    const bool hOK = (unsigned)hh < (unsigned)H_;
#pragma unroll
    for (int kc = 0; kc < 2; ++kc) {
      const int kk = bkk0 + kc;
      const bf16* xp = xs + (size_t)(b * CIN_ + i0 + kk) * HW_ + hh * W_;
      bf16* dst = &Bs[pp][bdh * 8 * LDS_STRIDE + kk];
      if (kw == 0 && hOK) {
        // interior tap column: 8 contiguous bf16, 16B-aligned -> one b128
        union { uint4 q; bf16 e[8]; } u;
        u.q = *(const uint4*)(xp + wBase);
#pragma unroll
        for (int j = 0; j < 8; ++j)
          dst[j * LDS_STRIDE] = (bf16)((float)u.e[j] * ak);
      } else {
#pragma unroll
        for (int j = 0; j < 8; ++j) {
          const int ww = wBase + j + kw;
          bf16 v = (bf16)0.f;
          if (hOK && (unsigned)ww < (unsigned)W_)
            v = (bf16)((float)xp[ww] * ak);
          dst[j * LDS_STRIDE] = v;
        }
      }
    }
  };

  const int lrow  = lane & 15;
  const int lhalf = lane >> 4;

  auto compute = [&](int pp) {
#pragma unroll
    for (int ks = 0; ks < 2; ++ks) {               // two 32-K WMMA sub-steps
      v16bf afrag[2], bfrag[2];
#pragma unroll
      for (int mi = 0; mi < 2; ++mi) {
        const int m = mWave + mi * 16 + lrow;
        union { uint4 q[2]; v16bf v; } u;
        // A: lanes 0-15 hold K{0-7,16-23}; lanes 16-31 hold K{8-15,24-31}
        u.q[0] = *(const uint4*)&As[pp][m * LDS_STRIDE + ks * 32 + lhalf * 8];
        u.q[1] = *(const uint4*)&As[pp][m * LDS_STRIDE + ks * 32 + lhalf * 8 + 16];
        afrag[mi] = u.v;
      }
#pragma unroll
      for (int ni = 0; ni < 2; ++ni) {
        const int n = nWave + ni * 16 + lrow;
        union { uint4 q[2]; v16bf v; } u;
        // B: lanes 0-15 hold K{0-7,16-23}; lanes 16-31 hold K{8-15,24-31} (column n)
        u.q[0] = *(const uint4*)&Bs[pp][n * LDS_STRIDE + ks * 32 + lhalf * 8];
        u.q[1] = *(const uint4*)&Bs[pp][n * LDS_STRIDE + ks * 32 + lhalf * 8 + 16];
        bfrag[ni] = u.v;
      }
#pragma unroll
      for (int mi = 0; mi < 2; ++mi)
#pragma unroll
        for (int ni = 0; ni < 2; ++ni)
          acc[mi][ni] = __builtin_amdgcn_wmma_f32_16x16x32_bf16(
              false, afrag[mi], false, bfrag[ni], (short)0, acc[mi][ni],
              false, false);
    }
  };

  // -------------------- double-buffered pipeline --------------------
  loadA(0, 0);
  loadB(0, 0);
  int p = 0;
  for (int kb = 0; kb < KDIM_; kb += KSTEP) {
    wait_asynccnt0();                              // buf[p] A tile landed
    __syncthreads();                               // buf[p] B stores visible,
                                                   // buf[1-p] readers done
    if (kb + KSTEP < KDIM_) {
      loadA(kb + KSTEP, 1 - p);                    // overlap with compute
      loadB(kb + KSTEP, 1 - p);
    }
    compute(p);
    p ^= 1;
  }

  // ---- epilogue: scale by attn_oup (f32) and store ----
  const int lhi = (lane >> 4) * 8;                 // C/D layout: hi lanes = M+8
#pragma unroll
  for (int mi = 0; mi < 2; ++mi) {
#pragma unroll
    for (int ni = 0; ni < 2; ++ni) {
      const int n   = nWave + ni * 16 + lrow;
      const int pix = (hBase + (n >> 3)) * W_ + (wBase + (n & 7));
#pragma unroll
      for (int r = 0; r < 8; ++r) {
        const int o = oBase + mWave + mi * 16 + r + lhi;
        out[(size_t)(b * COUT_ + o) * HW_ + pix] = acc[mi][ni][r] * attn_oup[b * COUT_ + o];
      }
    }
  }
}

// ----------------------------- launcher -------------------------------------
extern "C" void kernel_launch(void* const* d_in, const int* in_sizes, int n_in,
                              void* d_out, int out_size, void* d_ws, size_t ws_size,
                              hipStream_t stream) {
  const float* x       = (const float*)d_in[0];
  const float* w_share = (const float*)d_in[1];
  const float* b_share = (const float*)d_in[2];
  const float* w_inp   = (const float*)d_in[3];
  const float* b_inp   = (const float*)d_in[4];
  const float* w_oup   = (const float*)d_in[5];
  const float* b_oup   = (const float*)d_in[6];
  const float* w_k     = (const float*)d_in[7];
  const float* b_k     = (const float*)d_in[8];
  const float* weight  = (const float*)d_in[9];
  float* out = (float*)d_out;

  char* ws = (char*)d_ws;
  float* pooled  = (float*)(ws + 0);               //  32768 B
  float* ainp    = (float*)(ws + 32768);           //  32768 B
  float* aoup    = (float*)(ws + 65536);           //  32768 B
  float* ak      = (float*)(ws + 98304);           //   1152 B (pad 4096)
  bf16*  wTp     = (bf16*) (ws + 102400);          // 1179648 B
  bf16*  xsp     = (bf16*) (ws + 1282048);         // 51380224 B

  pool_kernel<<<B_ * CIN_, 256, 0, stream>>>(x, pooled);
  attn_kernel<<<1, 256, 0, stream>>>(pooled, w_share, b_share, w_inp, b_inp,
                                     w_oup, b_oup, w_k, b_k, ainp, aoup, ak);
  {
    const long long N4 = (long long)B_ * CIN_ * HW_ / 4;
    scale_x_kernel<<<(unsigned)((N4 + 255) / 256), 256, 0, stream>>>(x, ainp, xsp);
  }
  wT_kernel<<<(COUT_ * KDIM_ + 255) / 256, 256, 0, stream>>>(weight, wTp);

  dim3 grid(49, COUT_ / M_TILE, B_);
  dyconv_wmma_kernel<<<grid, 256, 0, stream>>>(xsp, wTp, ak, aoup, out);
}
